// MultiHeadGateAttentionLayer_85899345920380
// MI455X (gfx1250) — compile-verified, matrix operand order checked by reading, not run on hardware
//
#include <hip/hip_runtime.h>
#include <hip/hip_bf16.h>
#include <math.h>

// ---------------------------------------------------------------------------
// MultiHeadGateAttentionLayer for MI455X (gfx1250), bf16 WMMA pipeline with
// CDNA5 async global->LDS staging (ASYNCcnt) and double-buffered LDS tiles.
// o  = (gelu(u@Wu+bu) * softmax(rope_q @ rope_k^T / sqrt(E)) @ gelu(v@Wv+bv)) @ Wo + bo
// Outputs: o [B,L,D] fp32  ++  attn [B,H,L,S] fp32 (written in place in d_out).
// ---------------------------------------------------------------------------

typedef __bf16 bf16_t;
typedef __attribute__((ext_vector_type(16))) __bf16 v16bf;
typedef __attribute__((ext_vector_type(8)))  __bf16 bf8;
typedef __attribute__((ext_vector_type(8)))  float  v8f;

#define B_  4
#define L_  2048
#define S_  2048
#define D_  1024
#define H_  8
#define E_  128
#define UV_ 128
#define HE_ 1024   // H*E == H*UV

#define BM 128
#define BN 128
#define BK 32
#define LDSP (BK + 8)   // padded LDS inner dim (bf16); row pitch 80B (16B multiple)

__device__ __forceinline__ float gelu_erf(float x) {
  return 0.5f * x * (1.0f + erff(x * 0.70710678118654752f));
}

// CDNA5 async 16-byte copy: global memory -> LDS, tracked by ASYNCcnt.
// lds_off is the 32-bit LDS byte address (low 32 bits of the flat shared ptr).
__device__ __forceinline__ void async_b128(unsigned lds_off, const void* gptr) {
  asm volatile("global_load_async_to_lds_b128 %0, %1, off"
               :: "v"(lds_off), "v"((unsigned long long)(size_t)gptr)
               : "memory");
}
__device__ __forceinline__ void wait_async0() {
  asm volatile("s_wait_asynccnt 0x0" ::: "memory");
}

// ---------------------------------------------------------------------------
// fp32 -> bf16 elementwise conversion
// ---------------------------------------------------------------------------
__global__ void cvt_f32_bf16(const float* __restrict__ in, bf16_t* __restrict__ out, int n) {
  for (int i = blockIdx.x * blockDim.x + threadIdx.x; i < n; i += gridDim.x * blockDim.x)
    out[i] = (bf16_t)in[i];
}

// fp32 [Kd x Nd] row-major  ->  bf16 [Nd x Kd] row-major (weight pre-transpose
// so every GEMM B operand is N x K rows => async-copyable into LDS^T directly)
__global__ void cvt_transpose_bf16(const float* __restrict__ in, bf16_t* __restrict__ out,
                                   int Kd, int Nd) {
  int i = blockIdx.x * blockDim.x + threadIdx.x;
  if (i >= Kd * Nd) return;
  int k = i / Nd, n = i - k * Nd;
  out[(long long)n * Kd + k] = (bf16_t)in[i];
}

// v_bf [B,S,H*UV] -> vT [B,H,UV,S]  (bf16), so ctx GEMM B operand is N x K rows
__global__ void transpose_v(const bf16_t* __restrict__ in, bf16_t* __restrict__ out) {
  long long i = (long long)blockIdx.x * blockDim.x + threadIdx.x;
  if (i >= (long long)B_ * S_ * HE_) return;
  int c = (int)(i % HE_);
  long long row = i / HE_;              // b*S + s
  int b = (int)(row / S_), s = (int)(row - (long long)b * S_);
  int h = c / UV_, n = c - h * UV_;
  out[(((long long)(b * H_ + h) * UV_ + n) * S_) + s] = in[i];
}

// ---------------------------------------------------------------------------
// Flexible tiled WMMA GEMM:  C = epilogue( scale * A@B^T + bias )
//   A: bf16 (a_f32==0, async-staged) or fp32 (a_f32==1, converted on stage),
//      row-major M x K.
//   B: bf16, N x K row-major (ldb) -> rows copied straight into LDS^T (async).
//   Batch (gridDim.z = B_*H_ or 1): off = (z/H_)*strideB + (z%H_)*strideH.
//   Epilogue: x = acc*scale + bias[n]; optional exact-erf GELU; optional fp32
//             gate multiply; store fp32 or bf16.
// 128x128 block tile, 8 waves (4x2), each wave 2x4 tiles of
// v_wmma_f32_16x16x32_bf16; double-buffered LDS with async global->LDS copies.
// ---------------------------------------------------------------------------
__global__ __launch_bounds__(256) void gemm_wmma(
    const void* __restrict__ Av, int a_f32, long long lda, long long sAb, long long sAh,
    const bf16_t* __restrict__ Bm, long long ldb, long long sBb, long long sBh,
    const float* __restrict__ bias,
    const float* __restrict__ gate, long long ldg, long long sGb, long long sGh,
    void* __restrict__ Cv, int c_f32, long long ldc, long long sCb, long long sCh,
    int M, int N, int K, int do_gelu, float scale)
{
  __shared__ bf16_t As[2][BM][LDSP];    // 2 x 10 KB
  __shared__ bf16_t BsT[2][BN][LDSP];   // 2 x 10 KB

  const int z  = blockIdx.z;
  const int zb = z / H_, zh = z % H_;
  const long long aoff = (long long)zb * sAb + (long long)zh * sAh;
  const long long boff = (long long)zb * sBb + (long long)zh * sBh;
  const long long coff = (long long)zb * sCb + (long long)zh * sCh;
  const long long goff = (long long)zb * sGb + (long long)zh * sGh;

  const int m0   = blockIdx.x * BM;
  const int n0   = blockIdx.y * BN;
  const int tid  = threadIdx.x;
  const int wave = tid >> 5;
  const int lane = tid & 31;
  const int wm   = wave >> 1;      // 0..3 : 32-row strip
  const int wn   = wave & 1;       // 0..1 : 64-col strip
  const int lrow = lane & 15;
  const int lhalf = lane >> 4;
  const int offK  = lhalf * 8;     // ISA 16-bit A/B lane swizzle: K base 0 or 8 (+16)

  const bf16_t* Ab = (const bf16_t*)Av;
  const float*  Af = (const float*)Av;

  v8f acc[2][4];
#pragma unroll
  for (int mi = 0; mi < 2; ++mi)
#pragma unroll
    for (int ni = 0; ni < 4; ++ni)
#pragma unroll
      for (int r = 0; r < 8; ++r) acc[mi][ni][r] = 0.0f;

  // ---- tile staging (issues async copies; fp32-A path stores synchronously) ----
  auto stage = [&](int buf, int k0) {
    // A tile: 128 x 32 bf16 = 512 chunks of 16B, 2 per thread
    for (int c = tid; c < BM * (BK / 8); c += 256) {
      int r = c >> 2, seg = (c & 3) * 8;
      long long g = aoff + (long long)(m0 + r) * lda + k0 + seg;
      if (a_f32) {
#pragma unroll
        for (int j = 0; j < 8; ++j) As[buf][r][seg + j] = (bf16_t)Af[g + j];
      } else {
        async_b128((unsigned)(size_t)&As[buf][r][seg], &Ab[g]);
      }
    }
    // B tile rows (N x K layout): direct 16B async copies into LDS^T
    for (int c = tid; c < BN * (BK / 8); c += 256) {
      int nr = c >> 2, seg = (c & 3) * 8;
      async_b128((unsigned)(size_t)&BsT[buf][nr][seg],
                 &Bm[boff + (long long)(n0 + nr) * ldb + k0 + seg]);
    }
  };

  const int nk = K / BK;
  stage(0, 0);                               // prologue: fill buffer 0

  for (int ki = 0; ki < nk; ++ki) {
    const int cur = ki & 1;
    wait_async0();                           // our async copies into `cur` done
    __syncthreads();                         // all waves: cur full, prev reads done
    if (ki + 1 < nk) stage(cur ^ 1, (ki + 1) * BK);   // overlap next fetch w/ compute

    // ---- fragments per ISA layout: lane<16 K{0..7,16..23}, lane>=16 K{8..15,24..31} ----
    v16bf afrag[2], bfrag[4];
#pragma unroll
    for (int mi = 0; mi < 2; ++mi) {
      int r = wm * 32 + mi * 16 + lrow;
      bf8 lo = *(const bf8*)&As[cur][r][offK];
      bf8 hi = *(const bf8*)&As[cur][r][offK + 16];
#pragma unroll
      for (int j = 0; j < 8; ++j) { afrag[mi][j] = lo[j]; afrag[mi][8 + j] = hi[j]; }
    }
#pragma unroll
    for (int ni = 0; ni < 4; ++ni) {
      int cc = wn * 64 + ni * 16 + lrow;
      bf8 lo = *(const bf8*)&BsT[cur][cc][offK];
      bf8 hi = *(const bf8*)&BsT[cur][cc][offK + 16];
#pragma unroll
      for (int j = 0; j < 8; ++j) { bfrag[ni][j] = lo[j]; bfrag[ni][8 + j] = hi[j]; }
    }

#pragma unroll
    for (int mi = 0; mi < 2; ++mi)
#pragma unroll
      for (int ni = 0; ni < 4; ++ni)
        acc[mi][ni] = __builtin_amdgcn_wmma_f32_16x16x32_bf16(
            false, afrag[mi], false, bfrag[ni], (short)0, acc[mi][ni], false, false);
  }

  // ---- epilogue: C lane map: M = r + 8*lhalf, N = lrow within each 16x16 tile ----
#pragma unroll
  for (int mi = 0; mi < 2; ++mi) {
#pragma unroll
    for (int ni = 0; ni < 4; ++ni) {
      int ncol  = n0 + wn * 64 + ni * 16 + lrow;
      float bv  = bias ? bias[ncol] : 0.0f;
      int mbase = m0 + wm * 32 + mi * 16 + lhalf * 8;
#pragma unroll
      for (int r = 0; r < 8; ++r) {
        long long mrow = mbase + r;
        float x = acc[mi][ni][r] * scale + bv;
        if (do_gelu) x = gelu_erf(x);
        if (gate) x *= gate[goff + mrow * ldg + ncol];
        if (c_f32) ((float*)Cv)[coff + mrow * ldc + ncol] = x;
        else       ((bf16_t*)Cv)[coff + mrow * ldc + ncol] = (bf16_t)x;
      }
    }
  }
}

// ---------------------------------------------------------------------------
// ScaleOffset + interleaved-pair RoPE on flattened HE dim; emit bf16 q and k.
// One thread per pair (n0 = 2*pp). freq = t * 10000^(-n0/HE).
// ---------------------------------------------------------------------------
__global__ void rope_affine(const float* __restrict__ qk,
                            const float* __restrict__ qg, const float* __restrict__ qb,
                            const float* __restrict__ kg, const float* __restrict__ kb,
                            bf16_t* __restrict__ qo, bf16_t* __restrict__ ko)
{
  const int NP = HE_ / 2;                       // 512 pairs per row
  int p = blockIdx.x * blockDim.x + threadIdx.x;
  if (p >= B_ * L_ * NP) return;
  int row = p / NP;                             // 0 .. B*L-1
  int pp  = p - row * NP;
  int t   = row & (L_ - 1);
  int n0  = pp * 2;
  int e0  = n0 & (E_ - 1);

  long long base = (long long)row * HE_ + n0;
  float x0 = qk[base], x1 = qk[base + 1];

  float fr = (float)t * powf(10000.0f, -(float)n0 / (float)HE_);
  float s, c;
  sincosf(fr, &s, &c);

  float q0 = x0 * qg[e0] + qb[e0], q1 = x1 * qg[e0 + 1] + qb[e0 + 1];
  float k0 = x0 * kg[e0] + kb[e0], k1 = x1 * kg[e0 + 1] + kb[e0 + 1];

  qo[base]     = (bf16_t)(q0 * c - q1 * s);
  qo[base + 1] = (bf16_t)(q1 * c + q0 * s);
  ko[base]     = (bf16_t)(k0 * c - k1 * s);
  ko[base + 1] = (bf16_t)(k1 * c + k0 * s);
}

// ---------------------------------------------------------------------------
// In-place row softmax over S=2048 (one 256-thread block per row, 8 elems/thread)
// ---------------------------------------------------------------------------
__global__ __launch_bounds__(256) void softmax_rows(float* __restrict__ attn)
{
  __shared__ float red[256];
  float* p = attn + (long long)blockIdx.x * S_;
  int tid = threadIdx.x;

  float v[8], vmax = -3.402823466e38f;
#pragma unroll
  for (int i = 0; i < 8; ++i) { v[i] = p[tid + i * 256]; vmax = fmaxf(vmax, v[i]); }
  red[tid] = vmax; __syncthreads();
  for (int s = 128; s > 0; s >>= 1) {
    if (tid < s) red[tid] = fmaxf(red[tid], red[tid + s]);
    __syncthreads();
  }
  vmax = red[0]; __syncthreads();

  float sum = 0.0f;
#pragma unroll
  for (int i = 0; i < 8; ++i) { v[i] = expf(v[i] - vmax); sum += v[i]; }
  red[tid] = sum; __syncthreads();
  for (int s = 128; s > 0; s >>= 1) {
    if (tid < s) red[tid] += red[tid + s];
    __syncthreads();
  }
  float inv = 1.0f / red[0];
#pragma unroll
  for (int i = 0; i < 8; ++i) p[tid + i * 256] = v[i] * inv;
}

// ---------------------------------------------------------------------------
extern "C" void kernel_launch(void* const* d_in, const int* in_sizes, int n_in,
                              void* d_out, int out_size, void* d_ws, size_t ws_size,
                              hipStream_t stream) {
  const float* u_in = (const float*)d_in[0];
  const float* qry  = (const float*)d_in[1];
  // d_in[2] ("keys") is unused by the reference: k derives from queries@Wqk.
  const float* val  = (const float*)d_in[3];
  const float* Wqk  = (const float*)d_in[4];
  const float* bqk  = (const float*)d_in[5];
  const float* Wv   = (const float*)d_in[6];
  const float* bv   = (const float*)d_in[7];
  const float* Wu   = (const float*)d_in[8];
  const float* bu   = (const float*)d_in[9];
  const float* Wo   = (const float*)d_in[10];
  const float* bo   = (const float*)d_in[11];
  const float* qg   = (const float*)d_in[12];
  const float* qb   = (const float*)d_in[13];
  const float* kg   = (const float*)d_in[14];
  const float* kb   = (const float*)d_in[15];

  float* o_out = (float*)d_out;
  float* attn  = o_out + (size_t)B_ * L_ * D_;   // attn region of d_out

  // ---- workspace carve-up (~136 MB with lifetime-based aliasing) ----
  char* ws = (char*)d_ws;
  size_t off = 0;
  auto take = [&](size_t bytes) -> char* {
    char* p = ws + off;
    off += (bytes + 255) & ~(size_t)255;
    return p;
  };
  const size_t MDbf = (size_t)B_ * L_ * D_ * sizeof(bf16_t);   // 16 MB
  const size_t Wbf  = (size_t)D_ * HE_ * sizeof(bf16_t);       //  2 MB
  const size_t MDf  = (size_t)B_ * L_ * HE_ * sizeof(float);   // 32 MB

  bf16_t* qry_bf = (bf16_t*)take(MDbf);
  bf16_t* val_bf = (bf16_t*)take(MDbf);
  bf16_t* u_bf   = (bf16_t*)take(MDbf);
  bf16_t* wqkT   = (bf16_t*)take(Wbf);   // [HE][D]  (N x K)
  bf16_t* wvT    = (bf16_t*)take(Wbf);   // [HE][D]
  bf16_t* wuT    = (bf16_t*)take(Wbf);   // [HE][D]
  bf16_t* woT    = (bf16_t*)take(Wbf);   // [D][HE]
  float*  qk_f   = (float*)take(MDf);
  float*  up_f   = (float*)take(MDf);
  bf16_t* v_bf   = (bf16_t*)take(MDbf);
  // aliases (lifetimes disjoint in the launch order below):
  bf16_t* q_bf  = u_bf;             // u_bf dead after u_p GEMM
  bf16_t* k_bf  = val_bf;           // val_bf dead after v GEMM (v_bf holds v)
  bf16_t* gated = qry_bf;           // qry_bf dead after qk GEMM
  bf16_t* vT    = (bf16_t*)qk_f;    // qk_f dead after rope; vT is 16 MB <= 32 MB

  const int nMD = B_ * L_ * D_;       // 8388608
  const int nW  = D_ * HE_;           // 1048576
  dim3 blk(256);

  // 1) bf16 conversions + weight pre-transposes (N x K for async B staging)
  cvt_f32_bf16<<<4096, blk, 0, stream>>>(qry,  qry_bf, nMD);
  cvt_f32_bf16<<<4096, blk, 0, stream>>>(val,  val_bf, nMD);
  cvt_f32_bf16<<<4096, blk, 0, stream>>>(u_in, u_bf,   nMD);
  cvt_transpose_bf16<<<nW / 256, blk, 0, stream>>>(Wqk, wqkT, D_, HE_);
  cvt_transpose_bf16<<<nW / 256, blk, 0, stream>>>(Wv,  wvT,  D_, HE_);
  cvt_transpose_bf16<<<nW / 256, blk, 0, stream>>>(Wu,  wuT,  D_, HE_);
  cvt_transpose_bf16<<<nW / 256, blk, 0, stream>>>(Wo,  woT,  HE_, D_);

  dim3 gproj(B_ * L_ / BM, HE_ / BN, 1);   // 64 x 8

  // 2) qk = gelu(queries @ Wqk + bqk)  -> fp32
  gemm_wmma<<<gproj, blk, 0, stream>>>(qry_bf, 0, D_, 0, 0,
                                       wqkT, D_, 0, 0,
                                       bqk, nullptr, 0, 0, 0,
                                       qk_f, 1, HE_, 0, 0,
                                       B_ * L_, HE_, D_, 1, 1.0f);
  // 3) u_p = gelu(u @ Wu + bu) -> fp32 (gate)
  gemm_wmma<<<gproj, blk, 0, stream>>>(u_bf, 0, D_, 0, 0,
                                       wuT, D_, 0, 0,
                                       bu, nullptr, 0, 0, 0,
                                       up_f, 1, HE_, 0, 0,
                                       B_ * L_, HE_, D_, 1, 1.0f);
  // 4) v = gelu(values @ Wv + bv) -> bf16
  gemm_wmma<<<gproj, blk, 0, stream>>>(val_bf, 0, D_, 0, 0,
                                       wvT, D_, 0, 0,
                                       bv, nullptr, 0, 0, 0,
                                       v_bf, 0, HE_, 0, 0,
                                       B_ * L_, HE_, D_, 1, 1.0f);
  // 5) affine + RoPE -> q_bf, k_bf (bf16)   [qk_f becomes dead after this]
  {
    int npairs = B_ * L_ * (HE_ / 2);
    rope_affine<<<(npairs + 255) / 256, blk, 0, stream>>>(qk_f, qg, qb, kg, kb, q_bf, k_bf);
  }
  // 5b) v_bf [B,S,HE] -> vT [B,H,UV,S]  (N x K rows for ctx GEMM, async-stageable)
  transpose_v<<<nMD / 256, blk, 0, stream>>>(v_bf, vT);
  // 6) scores = q @ k^T / sqrt(E)  -> attn region (raw, fp32)
  {
    dim3 g(L_ / BM, S_ / BN, B_ * H_);   // 16 x 16 x 32
    gemm_wmma<<<g, blk, 0, stream>>>(q_bf, 0, HE_, (long long)L_ * HE_, E_,
                                     k_bf, HE_, (long long)L_ * HE_, E_,
                                     nullptr, nullptr, 0, 0, 0,
                                     attn, 1, S_, (long long)H_ * L_ * S_, (long long)L_ * S_,
                                     L_, S_, E_, 0, 0.08838834764831845f);
  }
  // 7) softmax in place on attn
  softmax_rows<<<B_ * H_ * L_, blk, 0, stream>>>(attn);
  // 8) gated ctx = (attn @ v) * u_p -> bf16
  {
    dim3 g(L_ / BM, UV_ / BN, B_ * H_);  // 16 x 1 x 32
    gemm_wmma<<<g, blk, 0, stream>>>(attn, 1, S_, (long long)H_ * L_ * S_, (long long)L_ * S_,
                                     vT, S_, (long long)H_ * UV_ * S_, (long long)UV_ * S_,
                                     nullptr,
                                     up_f, HE_, (long long)L_ * HE_, UV_,
                                     gated, 0, HE_, (long long)L_ * HE_, UV_,
                                     L_, UV_, S_, 0, 1.0f);
  }
  // 9) o = gated @ Wo + bo -> fp32 d_out
  gemm_wmma<<<gproj, blk, 0, stream>>>(gated, 0, HE_, 0, 0,
                                       woT, HE_, 0, 0,
                                       bo, nullptr, 0, 0, 0,
                                       o_out, 1, D_, 0, 0,
                                       B_ * L_, D_, HE_, 0, 1.0f);
}
